// SemiSupervisedNetwork_23381801959994
// MI455X (gfx1250) — matrix operand order repaired and checked by reading, not compiled, not run on hardware
//
#include <hip/hip_runtime.h>

// ---------------------------------------------------------------------------
// LIF SNN forward (B=128, NIN=784, NHID=1024, NOUT=10, T=200) for gfx1250.
//
//   P1: pack input spikes -> f16, rows (t*B+b) x KPAD (KPAD=800, zero pad)
//   P2: pack relu(W1)^T -> f16 (NHID x KPAD)    [WMMA B rows, K-contiguous]
//   P3: pack relu(W2)^T -> f16 (16 x NHID)      [N_OUT padded to 16]
//   K1: fused hidden GEMM + LIF scan. Wave tile 16(batch)x16(hid); v_h lives
//       in 8 fp32 VGPRs across t. Per t: 25x v_wmma_f32_16x16x32_f16 K-chain.
//       A tile (16x800 f16) is DMA'd to LDS with GLOBAL_LOAD_ASYNC_TO_LDS_B128,
//       double-buffered (ASYNCcnt), shared by the 4 waves. B (W1t) stays
//       register-resident across all t (compiler hoists ~200 VGPRs).
//   K2: output GEMM I_o[t] = s_h[t-1] @ W2 as one 25600x1024x16 WMMA GEMM.
//   K3: output LIF scan + firing rates.
// ---------------------------------------------------------------------------

typedef _Float16 f16;
typedef _Float16 v8h  __attribute__((ext_vector_type(8)));
typedef _Float16 v16h __attribute__((ext_vector_type(16)));
typedef float    v8f  __attribute__((ext_vector_type(8)));
typedef int      v4i  __attribute__((ext_vector_type(4)));

typedef __attribute__((address_space(1))) v4i v4i_g;   // global int4
typedef __attribute__((address_space(3))) v4i v4i_l;   // LDS int4

#define B_     128
#define NIN    784
#define NHID   1024
#define NOUT   10
#define T_     200
#define KPAD   800            // 784 padded up to 25 * 32
#define KC1    25             // KPAD / 32
#define KC2    32             // NHID / 32
#define DECAY  0.95f          // 1 - dt/tau
#define VTH    1.0f

#define ROWB    1600          // bytes per A row (KPAD * 2)
#define LROWB   1664          // LDS row stride (padded: 104 x 16B chunks)
#define LROWH   832           // LDS row stride in halfs
#define TSTEPB  (B_ * ROWB)   // 204800 B: global advance of A tile per timestep

// workspace byte offsets (all 64B aligned)
#define OFF_XH   0ull                              // f16 [T*B][KPAD]   40,960,000 B
#define OFF_W1T  40960000ull                       // f16 [NHID][KPAD]   1,638,400 B
#define OFF_W2T  42598400ull                       // f16 [16][NHID]        32,768 B
#define OFF_SH   42631168ull                       // f16 [T*B][NHID]   52,428,800 B
#define OFF_IO   95059968ull                       // f32 [T*B][16]      1,638,400 B

// ---- CDNA5 async copy global->LDS (ASYNCcnt) ------------------------------
// Builtin signature (from toolchain diagnostics): (int4 AS1*, int4 AS3*,
// imm offset, imm cpol) — non-const pointers. Built via inttoptr casts.
#if __has_builtin(__builtin_amdgcn_global_load_async_to_lds_b128)
#define ASYNC_CP16(gp, lp)                                                \
    __builtin_amdgcn_global_load_async_to_lds_b128(                       \
        (v4i_g*)(unsigned long long)(size_t)(gp),                         \
        (v4i_l*)(unsigned)(size_t)(lp), 0, 0)
#else
#define ASYNC_CP16(gp, lp)                                                \
    asm volatile("global_load_async_to_lds_b128 %0, %1, off" ::           \
                 "v"((unsigned)(size_t)(lp)),                             \
                 "v"((unsigned long long)(size_t)(gp)) : "memory")
#endif

#if __has_builtin(__builtin_amdgcn_s_wait_asynccnt)
#define WAIT_ASYNC(n) __builtin_amdgcn_s_wait_asynccnt(n)
#else
#define WAIT_ASYNC(n) asm volatile("s_wait_asynccnt %0" :: "i"(n) : "memory")
#endif

// ---------------- pre-pass packing kernels ----------------

__global__ void pack_x(const float* __restrict__ in, f16* __restrict__ xh) {
    int idx = blockIdx.x * blockDim.x + threadIdx.x;     // exact: T*B*KPAD threads
    int k  = idx % KPAD;
    int tb = idx / KPAD;
    int b  = tb % B_;
    int t  = tb / B_;
    float v = (k < NIN) ? in[b * (NIN * T_) + k * T_ + t] : 0.0f;
    xh[idx] = (f16)v;
}

__global__ void pack_w1(const float* __restrict__ w1, f16* __restrict__ w1t) {
    int idx = blockIdx.x * blockDim.x + threadIdx.x;     // exact: NHID*KPAD threads
    int k = idx % KPAD;
    int n = idx / KPAD;
    float v = (k < NIN) ? w1[k * NHID + n] : 0.0f;
    w1t[idx] = (f16)(v > 0.0f ? v : 0.0f);               // relu folded into pack
}

__global__ void pack_w2(const float* __restrict__ w2, f16* __restrict__ w2t) {
    int idx = blockIdx.x * blockDim.x + threadIdx.x;     // exact: 16*NHID threads
    int k = idx % NHID;
    int n = idx / NHID;
    float v = (n < NOUT) ? w2[k * NOUT + n] : 0.0f;
    w2t[idx] = (f16)(v > 0.0f ? v : 0.0f);
}

// ---------------- K1: fused hidden GEMM + LIF scan ----------------
// grid (8, 16), block 128 (4 waves). Wave tile: 16 batch rows x 16 hidden cols.
// A tile double-buffered in LDS via async DMA; B register-resident across t.

__global__ __launch_bounds__(128) void snn_hidden(
    const f16* __restrict__ xh, const f16* __restrict__ w1t,
    float* __restrict__ hid_spk, f16* __restrict__ sh)
{
    __shared__ __align__(64) f16 atile[2][16 * LROWH];   // 2 x 26,624 B

    const int tid   = threadIdx.x;
    const int lane  = tid & 31;
    const int wave  = tid >> 5;
    const int m     = lane & 15;          // A row / B col / C col index
    const int half  = lane >> 4;          // K-half (A/B), M-half (C)
    const int bbase = blockIdx.x * 16;                  // batch tile base
    const int n0    = blockIdx.y * 64 + wave * 16;      // hidden tile base
    const int h     = n0 + m;

    // per-thread async copy plan: 13 x 16B chunks, invariant across t
    unsigned ldsoff[13];
    size_t   goff[13];
    const char* __restrict__ xb = (const char*)xh;
    #pragma unroll
    for (int i = 0; i < 13; ++i) {
        int c   = tid + i * 128;          // 0..1663
        int row = c / 104;                // 0..15
        int pos = c - row * 104;          // 0..103 (pos>=100 over-reads row pad)
        ldsoff[i] = (unsigned)(row * LROWB + pos * 16);
        goff[i]   = (size_t)(bbase + row) * ROWB + (size_t)pos * 16;
    }

    // B-matrix row pointer: column (n0+m) of relu(W1), contiguous in K
    const f16* __restrict__ pb = w1t + (size_t)(n0 + m) * KPAD;

    v8f vh = {0.f, 0.f, 0.f, 0.f, 0.f, 0.f, 0.f, 0.f};  // membrane potential

    // prologue: stream A(0) into buffer 0
    {
        char* lb = (char*)&atile[0][0];
        #pragma unroll
        for (int i = 0; i < 13; ++i)
            ASYNC_CP16(xb + goff[i], lb + ldsoff[i]);
    }

    for (int t = 0; t < T_; ++t) {
        const int buf = t & 1;
        if (t + 1 < T_) {                 // stream A(t+1) into the other buffer
            const char* gb = xb + (size_t)(t + 1) * TSTEPB;
            char* lb = (char*)&atile[buf ^ 1][0];
            #pragma unroll
            for (int i = 0; i < 13; ++i)
                ASYNC_CP16(gb + goff[i], lb + ldsoff[i]);
            WAIT_ASYNC(13);               // A(t) done (in-order completion)
        } else {
            WAIT_ASYNC(0);
        }
        __syncthreads();                  // A(t) visible to all 4 waves

        const f16* ap = &atile[buf][m * LROWH];
        v8f acc = {0.f, 0.f, 0.f, 0.f, 0.f, 0.f, 0.f, 0.f};
        #pragma unroll
        for (int kc = 0; kc < KC1; ++kc) {
            const int k0 = kc * 32;
            // A 16x32 f16 lane layout: two 16B chunks (ISA 7.12.2), from LDS
            v8h alo = *(const v8h*)(ap + k0 + half * 8);
            v8h ahi = *(const v8h*)(ap + k0 + 16 + half * 8);
            v16h a = __builtin_shufflevector(alo, ahi,
                0,1,2,3,4,5,6,7,8,9,10,11,12,13,14,15);
            // B 32x16 f16 lane layout: two 16B chunks, same K order as A (registers)
            v8h blo = *(const v8h*)(pb + k0 + half * 8);
            v8h bhi = *(const v8h*)(pb + k0 + 16 + half * 8);
            v16h b = __builtin_shufflevector(blo, bhi,
                0,1,2,3,4,5,6,7,8,9,10,11,12,13,14,15);
            acc = __builtin_amdgcn_wmma_f32_16x16x32_f16(
                false, a, false, b, (short)0, acc, false, false);
        }
        // LIF update: v = v*decay + I ; spike & reset. C layout: M = r + 8*half.
        #pragma unroll
        for (int r = 0; r < 8; ++r) {
            const int brow = bbase + r + 8 * half;
            float v = vh[r] * DECAY + acc[r];
            const bool fire = (v >= VTH);
            const float s = fire ? 1.0f : 0.0f;
            vh[r] = fire ? 0.0f : v;
            hid_spk[(size_t)brow * (NHID * T_) + (size_t)h * T_ + t] = s;
            sh[(size_t)(t * B_ + brow) * NHID + h] = (f16)s;
        }
        __syncthreads();                  // done reading buf before overwrite
    }
}

// ---------------- K2: output GEMM  Io[t*B+b][n] = s_h[t-1] @ W2 ----------------
// 1600 wave-tiles of 16 rows; rows with t==0 get zeros (s_h(-1)=0).

__global__ __launch_bounds__(128) void out_gemm(
    const f16* __restrict__ sh, const f16* __restrict__ w2t,
    float* __restrict__ io)
{
    const int lane = threadIdx.x & 31;
    const int wave = threadIdx.x >> 5;
    const int m    = lane & 15;
    const int half = lane >> 4;
    const int rowbase = (blockIdx.x * 4 + wave) * 16;   // flattened (t*B+b) tile
    const int t = rowbase >> 7;                          // / B_ (128)

    if (t == 0) {
        #pragma unroll
        for (int r = 0; r < 8; ++r)
            io[(size_t)(rowbase + r + 8 * half) * 16 + m] = 0.0f;
        return;
    }
    const f16* __restrict__ pa = sh + (size_t)(rowbase - B_ + m) * NHID; // t-1 shift
    const f16* __restrict__ pb = w2t + (size_t)m * NHID;
    v8f acc = {0.f, 0.f, 0.f, 0.f, 0.f, 0.f, 0.f, 0.f};
    #pragma unroll
    for (int kc = 0; kc < KC2; ++kc) {
        const int k0 = kc * 32;
        v8h alo = *(const v8h*)(pa + k0 + half * 8);
        v8h ahi = *(const v8h*)(pa + k0 + 16 + half * 8);
        v16h a = __builtin_shufflevector(alo, ahi,
            0,1,2,3,4,5,6,7,8,9,10,11,12,13,14,15);
        v8h blo = *(const v8h*)(pb + k0 + half * 8);
        v8h bhi = *(const v8h*)(pb + k0 + 16 + half * 8);
        v16h b = __builtin_shufflevector(blo, bhi,
            0,1,2,3,4,5,6,7,8,9,10,11,12,13,14,15);
        acc = __builtin_amdgcn_wmma_f32_16x16x32_f16(
            false, a, false, b, (short)0, acc, false, false);
    }
    #pragma unroll
    for (int r = 0; r < 8; ++r)
        io[(size_t)(rowbase + r + 8 * half) * 16 + m] = acc[r];
}

// ---------------- K3: output LIF scan + firing rates ----------------

__global__ void out_lif(const float* __restrict__ io,
                        float* __restrict__ out_spk,
                        float* __restrict__ rates)
{
    int tid = blockIdx.x * blockDim.x + threadIdx.x;
    if (tid >= B_ * NOUT) return;
    const int b = tid / NOUT;
    const int o = tid % NOUT;
    float v = 0.0f, rate = 0.0f;
    for (int t = 0; t < T_; ++t) {
        float I = io[(size_t)(t * B_ + b) * 16 + o];
        v = v * DECAY + I;
        const bool fire = (v >= VTH);
        const float s = fire ? 1.0f : 0.0f;
        v = fire ? 0.0f : v;
        out_spk[(size_t)b * (NOUT * T_) + (size_t)o * T_ + t] = s;
        rate += s;
    }
    rates[b * NOUT + o] = rate * (1.0f / T_);
}

// ---------------- launcher ----------------

extern "C" void kernel_launch(void* const* d_in, const int* in_sizes, int n_in,
                              void* d_out, int out_size, void* d_ws, size_t ws_size,
                              hipStream_t stream)
{
    const float* in_spk = (const float*)d_in[0];   // (B, NIN, T)
    const float* w1     = (const float*)d_in[1];   // (NIN, NHID)
    const float* w2     = (const float*)d_in[2];   // (NHID, NOUT)

    char* ws = (char*)d_ws;
    f16*   xh  = (f16*)(ws + OFF_XH);
    f16*   w1t = (f16*)(ws + OFF_W1T);
    f16*   w2t = (f16*)(ws + OFF_W2T);
    f16*   sh  = (f16*)(ws + OFF_SH);
    float* io  = (float*)(ws + OFF_IO);

    float* hid_spk = (float*)d_out;                              // (B, NHID, T)
    float* out_spk = hid_spk + (size_t)B_ * NHID * T_;           // (B, NOUT, T)
    float* rates   = out_spk + (size_t)B_ * NOUT * T_;           // (B, NOUT)

    pack_x <<<(T_ * B_ * KPAD) / 256, 256, 0, stream>>>(in_spk, xh);
    pack_w1<<<(NHID * KPAD) / 256,    256, 0, stream>>>(w1, w1t);
    pack_w2<<<(16 * NHID) / 256,      256, 0, stream>>>(w2, w2t);

    dim3 g1(B_ / 16, NHID / 64);            // (8, 16), 4 waves each
    snn_hidden<<<g1, 128, 0, stream>>>(xh, w1t, hid_spk, sh);

    out_gemm<<<(T_ * B_ / 16) / 4, 128, 0, stream>>>(sh, w2t, io); // 400 WGs
    out_lif <<<(B_ * NOUT + 255) / 256, 256, 0, stream>>>(io, out_spk, rates);
}
